// bottleneckFusion_72808285602187
// MI455X (gfx1250) — matrix operand order repaired and checked by reading, not compiled, not run on hardware
//
#include <hip/hip_runtime.h>
#include <hip/hip_bf16.h>

typedef __bf16 bf16_t;
typedef __bf16 v16bf __attribute__((ext_vector_type(16)));
typedef __bf16 v8bf  __attribute__((ext_vector_type(8)));
typedef float  v8f   __attribute__((ext_vector_type(8)));
typedef float  f4    __attribute__((ext_vector_type(4)));

#define SEQ   16
#define CH    256
#define NLAT  8
#define NKEY  48          // 3*SEQ
#define ITERS 3
#define SCL   0.0625f     // C^-0.5 = 1/16

// ---- LDS layout (bytes, per block == per wave) -------------------------
// qkv  [48][256] bf16   :     0 .. 24576   (row-major concat l|a|v)
// qkvT [256][48] bf16   : 24576 .. 49152   (transposed; no K pad — masked)
// fus  [16][256] bf16   : 49152 .. 57344
// fusT [256][16] bf16   : 57344 .. 65536   (transposed; no K pad — masked)
// sc   [16][64]  f32    : 65536 .. 69632   (scores scratch)
// P    [16][64]  bf16   : 69632 .. 71680   (fused-attn probs, padded)
// P2   [3][16][32] bf16 : 71680 .. 74752   (cross-attn probs, padded)
#define SMEM_BYTES 74752

__device__ __forceinline__ v16bf cat16(v8bf a, v8bf b) {
  return __builtin_shufflevector(a, b, 0,1,2,3,4,5,6,7,8,9,10,11,12,13,14,15);
}

// A fragment: 16x32 bf16, rows striped per lane, K = k0 + hi*16 + [0..15] (contiguous)
__device__ __forceinline__ v16bf load_afrag(const bf16_t* m, int stride,
                                            int mrow, int k0, int lo, int hi) {
  const bf16_t* p = m + (mrow + lo) * stride + k0 + hi * 16;
  return cat16(*(const v8bf*)(p), *(const v8bf*)(p + 8));
}

// B fragment: 32x16 bf16. Lane holds column n = lo, K = k0 + hi*16 + [0..15],
// contiguous in the transposed source X[n][k].
__device__ __forceinline__ v16bf load_bfrag(const bf16_t* x, int stride,
                                            int n0, int k0, int lo, int hi) {
  const bf16_t* p = x + (n0 + lo) * stride + k0 + hi * 16;
  return cat16(*(const v8bf*)(p), *(const v8bf*)(p + 8));
}

// Same, but the hi-half of the wave (K = k0+16..k0+31) delivers zeros: lets us
// drop the zero-padded K columns from LDS (pad handled in registers).
__device__ __forceinline__ v16bf load_bfrag_maskhi(const bf16_t* x, int stride,
                                                   int n0, int k0, int lo, int hi) {
  v16bf v = load_bfrag(x, stride, n0, k0, lo, hi);
  v16bf z = {};
  return hi ? z : v;
}

__device__ __forceinline__ v8f wmma_bf16(v16bf a, v16bf b, v8f c) {
  return __builtin_amdgcn_wmma_f32_16x16x32_bf16(false, a, false, b,
                                                 (short)0, c, false, false);
}

__global__ void __launch_bounds__(32)
bottleneck_fusion_kernel(const float* __restrict__ gl,
                         const float* __restrict__ ga,
                         const float* __restrict__ gv,
                         const float* __restrict__ glat,
                         const float* __restrict__ gsl,
                         const float* __restrict__ gsa,
                         const float* __restrict__ gsv,
                         float* __restrict__ out) {
  extern __shared__ char smem[];
  bf16_t* qkv  = (bf16_t*)(smem);
  bf16_t* qkvT = (bf16_t*)(smem + 24576);
  bf16_t* fus  = (bf16_t*)(smem + 49152);
  bf16_t* fusT = (bf16_t*)(smem + 57344);
  float*  sc   = (float*) (smem + 65536);
  bf16_t* P    = (bf16_t*)(smem + 69632);
  bf16_t* P2   = (bf16_t*)(smem + 71680);

  const int lane = threadIdx.x;          // wave32: one wave per block
  const int lo = lane & 15, hi = lane >> 4;
  const int b = blockIdx.x;
  const float sl = gsl[0], sa = gsa[0], sv = gsv[0];

  // speculative prefetch of next batch's rows (global_prefetch_b8)
  if (b + 1 < (int)gridDim.x) {
    size_t nb = (size_t)(b + 1) * SEQ * CH;
    __builtin_prefetch(gl + nb, 0, 0);
    __builtin_prefetch(ga + nb, 0, 0);
    __builtin_prefetch(gv + nb, 0, 0);
  }

  // ---- latents: 8 loop-invariant A-fragments kept in VGPRs all kernel ----
  // lane row m = lo (valid m<8, rows 8..15 are zero padding)
  v16bf latA[8];
  {
    const float* src = glat + lo * CH;
    #pragma unroll
    for (int kk = 0; kk < 8; ++kk) {
      int k0 = kk * 32;
      v16bf a;
      #pragma unroll
      for (int j = 0; j < 8; ++j) {
        a[j]     = (lo < NLAT) ? (bf16_t)src[k0 + hi * 16 + j]     : (bf16_t)0.0f;
        a[8 + j] = (lo < NLAT) ? (bf16_t)src[k0 + hi * 16 + 8 + j] : (bf16_t)0.0f;
      }
      latA[kk] = a;
    }
  }

  // ---- stream l/a/v once (b128 non-temporal), keep bf16 row-major + T ----
  for (int m3 = 0; m3 < 3; ++m3) {
    const float* g = (m3 == 0 ? gl : (m3 == 1 ? ga : gv)) + (size_t)b * SEQ * CH;
    const f4* g4 = (const f4*)g;
    for (int i = lane; i < SEQ * CH / 4; i += 32) {
      f4 x = __builtin_nontemporal_load(&g4[i]);
      int e = i * 4, r = e >> 8, c = e & 255;
      int row = m3 * SEQ + r;
      union { bf16_t h[4]; unsigned long long u; } pk;
      pk.h[0] = (bf16_t)x[0]; pk.h[1] = (bf16_t)x[1];
      pk.h[2] = (bf16_t)x[2]; pk.h[3] = (bf16_t)x[3];
      *(unsigned long long*)&qkv[row * CH + c] = pk.u;   // 8B-aligned b64 store
      qkvT[(c + 0) * NKEY + row] = pk.h[0];
      qkvT[(c + 1) * NKEY + row] = pk.h[1];
      qkvT[(c + 2) * NKEY + row] = pk.h[2];
      qkvT[(c + 3) * NKEY + row] = pk.h[3];
    }
  }
  __syncthreads();

  for (int it = 0; it < ITERS; ++it) {
    // ===== fused scores: lat[16pad][256] @ concat^T -> sc[16][48] =====
    for (int t = 0; t < 3; ++t) {
      v8f acc = {};
      #pragma unroll
      for (int kk = 0; kk < 8; ++kk) {
        v16bf bb = load_bfrag(qkv, CH, t * 16, kk * 32, lo, hi);
        acc = wmma_bf16(latA[kk], bb, acc);
      }
      #pragma unroll
      for (int r = 0; r < 8; ++r)
        sc[(r + hi * 8) * 64 + t * 16 + lo] = acc[r] * SCL;
    }
    __syncthreads();
    // softmax over 48 keys (8 valid latent rows) -> P[16][64] (pad = 0)
    if (lane < 16) {
      int r = lane;
      if (r < NLAT) {
        float mx = -3.0e38f;
        for (int j = 0; j < NKEY; ++j) mx = fmaxf(mx, sc[r * 64 + j]);
        float sum = 0.0f;
        for (int j = 0; j < NKEY; ++j) sum += __expf(sc[r * 64 + j] - mx);
        float inv = 1.0f / sum;
        for (int j = 0; j < NKEY; ++j)
          P[r * 64 + j] = (bf16_t)(__expf(sc[r * 64 + j] - mx) * inv);
        for (int j = NKEY; j < 64; ++j) P[r * 64 + j] = (bf16_t)0.0f;
      } else {
        for (int j = 0; j < 64; ++j) P[r * 64 + j] = (bf16_t)0.0f;
      }
    }
    __syncthreads();
    // ===== fused = P[16][64] @ concat[48][256] (K pad masked) =====
    {
      v16bf aP0 = load_afrag(P, 64, 0, 0, lo, hi);    // hoisted: t-invariant
      v16bf aP1 = load_afrag(P, 64, 0, 32, lo, hi);
      for (int t = 0; t < 16; ++t) {
        v16bf b0 = load_bfrag(qkvT, NKEY, t * 16, 0, lo, hi);
        v16bf b1 = load_bfrag_maskhi(qkvT, NKEY, t * 16, 32, lo, hi); // K 48..63 = 0
        v8f acc = {};
        acc = wmma_bf16(aP0, b0, acc);
        acc = wmma_bf16(aP1, b1, acc);
        int ch = t * 16 + lo;
        #pragma unroll
        for (int r = 0; r < 8; ++r)
          fus[(r + hi * 8) * CH + ch] = (bf16_t)acc[r];
        #pragma unroll
        for (int rp = 0; rp < 4; ++rp) {  // packed transposed store (b32 pairs)
          union { bf16_t h[2]; unsigned int u; } pk;
          pk.h[0] = (bf16_t)acc[2 * rp];
          pk.h[1] = (bf16_t)acc[2 * rp + 1];
          *(unsigned int*)&fusT[ch * 16 + hi * 8 + 2 * rp] = pk.u;
        }
      }
    }
    __syncthreads();

    // ===== cross scores for l,a,v fused: each fus B-frag feeds 3 WMMAs =====
    {
      v8f a0 = {}, a1 = {}, a2 = {};
      #pragma unroll
      for (int kk = 0; kk < 8; ++kk) {
        int k0 = kk * 32;
        v16bf bb = load_bfrag(fus, CH, 0, k0, lo, hi);
        v16bf q0 = load_afrag(qkv, CH, 0, k0, lo, hi);
        v16bf q1 = load_afrag(qkv, CH, 16, k0, lo, hi);
        v16bf q2 = load_afrag(qkv, CH, 32, k0, lo, hi);
        a0 = wmma_bf16(q0, bb, a0);
        a1 = wmma_bf16(q1, bb, a1);
        a2 = wmma_bf16(q2, bb, a2);
      }
      #pragma unroll
      for (int r = 0; r < 8; ++r) {
        int rr = (r + hi * 8) * 64;
        sc[rr + 0  + lo] = a0[r] * SCL;
        sc[rr + 16 + lo] = a1[r] * SCL;
        sc[rr + 32 + lo] = a2[r] * SCL;
      }
    }
    __syncthreads();
    // softmax over the 8 valid latents, all 3 modalities -> P2[3][16][32]
    if (lane < 16) {
      int r = lane;
      for (int m3 = 0; m3 < 3; ++m3) {
        const float* s = &sc[r * 64 + m3 * 16];
        float mx = -3.0e38f;
        for (int j = 0; j < NLAT; ++j) mx = fmaxf(mx, s[j]);
        float sum = 0.0f;
        for (int j = 0; j < NLAT; ++j) sum += __expf(s[j] - mx);
        float inv = 1.0f / sum;
        for (int j = 0; j < 32; ++j)
          P2[m3 * 512 + r * 32 + j] =
              (j < NLAT) ? (bf16_t)(__expf(s[j] - mx) * inv) : (bf16_t)0.0f;
      }
    }
    __syncthreads();
    // ===== out = P2 @ fused for all 3; each fusT B-frag feeds 3 WMMAs =====
    {
      v16bf aQ0 = load_afrag(P2,        32, 0, 0, lo, hi);  // hoisted
      v16bf aQ1 = load_afrag(P2 + 512,  32, 0, 0, lo, hi);
      v16bf aQ2 = load_afrag(P2 + 1024, 32, 0, 0, lo, hi);
      for (int t = 0; t < 16; ++t) {
        v16bf bb = load_bfrag_maskhi(fusT, 16, t * 16, 0, lo, hi); // K 16..31 = 0
        v8f o0 = {}, o1 = {}, o2 = {};
        o0 = wmma_bf16(aQ0, bb, o0);
        o1 = wmma_bf16(aQ1, bb, o1);
        o2 = wmma_bf16(aQ2, bb, o2);
        int ch = t * 16 + lo;
        #pragma unroll
        for (int m3 = 0; m3 < 3; ++m3) {
          const v8f& o = (m3 == 0 ? o0 : (m3 == 1 ? o1 : o2));
          const float s = (m3 == 0 ? sl : (m3 == 1 ? sa : sv));
          const int base = m3 * SEQ;
          bf16_t nh[8];
          #pragma unroll
          for (int r = 0; r < 8; ++r) {
            int row = base + r + hi * 8;
            float nv = fmaf(s, o[r], (float)qkv[row * CH + ch]);
            nh[r] = (bf16_t)nv;
            qkv[row * CH + ch] = nh[r];
          }
          #pragma unroll
          for (int rp = 0; rp < 4; ++rp) {  // packed transposed update
            union { bf16_t h[2]; unsigned int u; } pk;
            pk.h[0] = nh[2 * rp];
            pk.h[1] = nh[2 * rp + 1];
            *(unsigned int*)&qkvT[ch * NKEY + base + hi * 8 + 2 * rp] = pk.u;
          }
        }
      }
    }
    __syncthreads();
  }

  // ---- output: l1[:, -1, :] == qkv row 15 ----
  for (int c = lane; c < CH; c += 32)
    out[(size_t)b * CH + c] = (float)qkv[15 * CH + c];
}

extern "C" void kernel_launch(void* const* d_in, const int* in_sizes, int n_in,
                              void* d_out, int out_size, void* d_ws, size_t ws_size,
                              hipStream_t stream) {
  (void)n_in; (void)d_ws; (void)ws_size; (void)out_size;
  const float* gl   = (const float*)d_in[0];
  const float* ga   = (const float*)d_in[1];
  const float* gv   = (const float*)d_in[2];
  const float* glat = (const float*)d_in[3];
  const float* gsl  = (const float*)d_in[4];
  const float* gsa  = (const float*)d_in[5];
  const float* gsv  = (const float*)d_in[6];
  float* out = (float*)d_out;

  const int B = in_sizes[0] / (SEQ * CH);   // 8192
  bottleneck_fusion_kernel<<<dim3(B), dim3(32), SMEM_BYTES, stream>>>(
      gl, ga, gv, glat, gsl, gsa, gsv, out);
}
